// Yolo_64879775973833
// MI455X (gfx1250) — matrix operand order repaired and checked by reading, not compiled, not run on hardware
//
#include <hip/hip_runtime.h>
#include <hip/hip_bf16.h>

typedef __attribute__((ext_vector_type(16))) __bf16 v16bf;
typedef __attribute__((ext_vector_type(8)))  __bf16 v8bf;
typedef __attribute__((ext_vector_type(8)))  float  v8f;
typedef __attribute__((ext_vector_type(4)))  int    v4i;

#define TOT_ROWS 25200
#define NO_CH    85

// LDS layout: A tile 32 pos x 32 k bf16, W tile 256 o x 32 k bf16.
// Row stride padded to 80 bytes (16B-aligned chunks, spreads banks).
#define A_STRIDE_B 80
#define W_STRIDE_B 80
#define A_BYTES    (32 * A_STRIDE_B)
#define W_BYTES    (256 * W_STRIDE_B)

#if __has_builtin(__builtin_amdgcn_global_load_async_to_lds_b128)
#define HAVE_ASYNC_LDS 1
typedef __attribute__((address_space(1))) v4i* gp_v4i;
typedef __attribute__((address_space(3))) v4i* lp_v4i;
__device__ __forceinline__ void async_copy_b128(const void* g, void* l) {
  __builtin_amdgcn_global_load_async_to_lds_b128((gp_v4i)(g), (lp_v4i)(l), 0, 0);
}
#endif

// ---------------- weight prep: f32 [255,C] -> bf16 [256,C] (row 255 zero) ---
__global__ void prep_w(const float* __restrict__ w, __bf16* __restrict__ dst, int C) {
  int idx = blockIdx.x * blockDim.x + threadIdx.x;
  if (idx >= 256 * C) return;
  int o = idx / C;
  int c = idx - o * C;
  float v = (o < 255) ? w[o * C + c] : 0.0f;
  dst[idx] = (__bf16)v;
}

// ---------------- fused 1x1-conv GEMM (bf16 WMMA) + YOLO decode -------------
__launch_bounds__(256)
__global__ void yolo_head(const float* __restrict__ P,      // [16, C, HW] f32
                          const __bf16* __restrict__ Wb,    // [256, C] bf16
                          const float* __restrict__ bias,   // [255]
                          const float* __restrict__ anc6,   // [3,2] anchors (this scale)
                          float* __restrict__ out,          // [16, 25200, 85]
                          int C, int nx, int HW, float stride_px, int row_off) {
  __shared__ __align__(16) char smem[A_BYTES + W_BYTES];
  char* Alds = smem;
  char* Wlds = smem + A_BYTES;

  const int t    = threadIdx.x;
  const int lane = t & 31;
  const int wv   = t >> 5;
  const int m_off = (wv >> 2) * 16;   // wave's 16-position block within 32
  const int n_off = (wv & 3) * 64;    // wave's 64-channel block within 256

  const int pos0 = blockIdx.x * 32;   // 32 positions per block; HW % 32 == 0
  const int b    = pos0 / HW;         // uniform over block
  const int hw0  = pos0 - b * HW;
  const float* Pb = P + (size_t)b * C * HW + hw0;

  const float a0x = anc6[0], a0y = anc6[1];
  const float a1x = anc6[2], a1y = anc6[3];
  const float a2x = anc6[4], a2y = anc6[5];

  const int r15 = lane & 15;
  const int hiA = lane >> 4;          // 0 | 1
  const int a_k0 = hiA * 8;           // A-frag K origin (elements)
  const int b_k0 = hiA * 8;           // B-frag K origin (elements)

  // Accumulators seeded with bias: C/D layout has N = lane%16 for all 8 VGPRs.
  v8f acc[4];
#pragma unroll
  for (int j = 0; j < 4; ++j) {
    int o = n_off + j * 16 + r15;
    float bv = (o < 255) ? bias[o] : 0.0f;
#pragma unroll
    for (int e = 0; e < 8; ++e) acc[j][e] = bv;
  }

  const int KT = C >> 5;
  for (int kt = 0; kt < KT; ++kt) {
    const int c0 = kt << 5;
    __syncthreads();  // previous iter's LDS reads complete

    // ---- stage A: 32 pos x 32 c, f32 -> bf16, transposed to [m][k] ----
    {
      const int m  = t & 31;
      const int cp = t >> 5;  // 0..7
#pragma unroll
      for (int p = 0; p < 2; ++p) {
        const int cl = cp * 2 + p * 16;               // even, 0..30
        const float* src = Pb + (size_t)(c0 + cl) * HW + m;
        float f0 = src[0];
        float f1 = src[HW];
        union { __bf16 h[2]; unsigned u; } pk;
        pk.h[0] = (__bf16)f0;
        pk.h[1] = (__bf16)f1;
        *(unsigned*)(Alds + m * A_STRIDE_B + cl * 2) = pk.u;
      }
      if (kt + 1 < KT)  // prefetch next K-block of activations
        __builtin_prefetch(Pb + (size_t)(c0 + 32 + (t >> 5) * 2) * HW + (t & 31), 0, 1);
    }

    // ---- stage W: 256 o x 32 c bf16, row t -> LDS row t ----
    {
      const __bf16* src = Wb + (size_t)t * C + c0;
      char* dst = Wlds + t * W_STRIDE_B;
#ifdef HAVE_ASYNC_LDS
#pragma unroll
      for (int q = 0; q < 4; ++q) async_copy_b128(src + q * 8, dst + q * 16);
#else
#pragma unroll
      for (int q = 0; q < 4; ++q) *(v8bf*)(dst + q * 16) = *(const v8bf*)(src + q * 8);
#endif
    }
#ifdef HAVE_ASYNC_LDS
#if __has_builtin(__builtin_amdgcn_s_wait_asynccnt)
    __builtin_amdgcn_s_wait_asynccnt(0);
#endif
#endif
    __syncthreads();

    // ---- build fragments per ISA layouts, 4 WMMAs ----
    union Frag { v16bf v; v8bf h[2]; };
    Frag af;
    {
      // A 16x32 bf16: lane holds row M=lane%16; K chunks {k0..k0+7, k0+16..k0+23}
      const char* ap = Alds + (m_off + r15) * A_STRIDE_B + a_k0 * 2;
      af.h[0] = *(const v8bf*)(ap);
      af.h[1] = *(const v8bf*)(ap + 32);
    }
#pragma unroll
    for (int j = 0; j < 4; ++j) {
      // B 32x16 bf16: lane holds column N=lane%16; K chunks {k0..k0+7, k0+16..k0+23}
      Frag bf;
      const char* bp = Wlds + (n_off + j * 16 + r15) * W_STRIDE_B + b_k0 * 2;
      bf.h[0] = *(const v8bf*)(bp);
      bf.h[1] = *(const v8bf*)(bp + 32);
      acc[j] = __builtin_amdgcn_wmma_f32_16x16x32_bf16(
          false, af.v, false, bf.v, (short)0, acc[j], false, false);
    }
  }

  // ---- epilogue: sigmoid + box decode + coalesced store ----
#pragma unroll
  for (int j = 0; j < 4; ++j) {
    const int o = n_off + j * 16 + r15;
    if (o >= 255) continue;                       // padding channel
    const int a  = (o >= 170) ? 2 : (o >= 85 ? 1 : 0);
    const int ch = o - a * 85;
    const float ax = (a == 0) ? a0x : (a == 1 ? a1x : a2x);
    const float ay = (a == 0) ? a0y : (a == 1 ? a1y : a2y);
#pragma unroll
    for (int e = 0; e < 8; ++e) {
      const int m  = m_off + e + hiA * 8;         // D layout: M = e + 8*(lane>=16)
      const int hw = hw0 + m;
      const float s = 1.0f / (1.0f + __expf(-acc[j][e]));
      float v;
      if (ch == 0) {
        v = (2.0f * s + (float)(hw % nx) - 0.5f) * stride_px;
      } else if (ch == 1) {
        v = (2.0f * s + (float)(hw / nx) - 0.5f) * stride_px;
      } else if (ch == 2) {
        v = 4.0f * s * s * ax;
      } else if (ch == 3) {
        v = 4.0f * s * s * ay;
      } else {
        v = s;
      }
      const size_t row = (size_t)b * TOT_ROWS + row_off + (size_t)a * HW + hw;
      out[row * NO_CH + ch] = v;
    }
  }
}

extern "C" void kernel_launch(void* const* d_in, const int* in_sizes, int n_in,
                              void* d_out, int out_size, void* d_ws, size_t ws_size,
                              hipStream_t stream) {
  const float* p0 = (const float*)d_in[0];
  const float* p1 = (const float*)d_in[1];
  const float* p2 = (const float*)d_in[2];
  const float* w0 = (const float*)d_in[3];
  const float* b0 = (const float*)d_in[4];
  const float* w1 = (const float*)d_in[5];
  const float* b1 = (const float*)d_in[6];
  const float* w2 = (const float*)d_in[7];
  const float* b2 = (const float*)d_in[8];
  const float* anchors = (const float*)d_in[9];  // [3,3,2]
  float* out = (float*)d_out;

  // bf16 weight scratch: 256*(256+512+1024)*2 = 896 KB
  __bf16* W0 = (__bf16*)d_ws;
  __bf16* W1 = W0 + 256 * 256;
  __bf16* W2 = W1 + 256 * 512;

  prep_w<<<(256 * 256 + 255) / 256, 256, 0, stream>>>(w0, W0, 256);
  prep_w<<<(256 * 512 + 255) / 256, 256, 0, stream>>>(w1, W1, 512);
  prep_w<<<(256 * 1024 + 255) / 256, 256, 0, stream>>>(w2, W2, 1024);

  // scale 0: 16*6400 positions / 32 = 3200 blocks
  yolo_head<<<3200, 256, 0, stream>>>(p0, W0, b0, anchors + 0,  out,  256, 80, 6400,  8.0f, 0);
  // scale 1: 16*1600 / 32 = 800
  yolo_head<<<800,  256, 0, stream>>>(p1, W1, b1, anchors + 6,  out,  512, 40, 1600, 16.0f, 19200);
  // scale 2: 16*400 / 32 = 200
  yolo_head<<<200,  256, 0, stream>>>(p2, W2, b2, anchors + 12, out, 1024, 20, 400,  32.0f, 24000);
}
